// AttentionMap_5600637354231
// MI455X (gfx1250) — matrix operand order repaired and checked by reading, not compiled, not run on hardware
//
#include <hip/hip_runtime.h>

typedef __attribute__((ext_vector_type(16))) _Float16 v16h;
typedef __attribute__((ext_vector_type(8)))  float    v8f;
typedef __attribute__((ext_vector_type(4)))  float    v4f;

#define S_LEN   2048
#define DHEAD   64
#define PDHEAD  32
#define NBATCH  2
#define NHEAD   16
#define SCALE   0.125f               // 64^-0.5
#define PSCALE  0.17677669529663687f // 32^-0.5

__device__ __forceinline__ float clamp5(float x) {
    return fminf(fmaxf(x, -5.0f), 5.0f);
}

// A-matrix fragment (f16 16x32, M x K), contiguous k order (matches B):
//   lanes 0-15 : row M=lane,    K = kb+0..15
//   lanes 16-31: row M=lane-16, K = kb+16..31
__device__ __forceinline__ v16h load_frag_a(const float* __restrict__ row, int kb, int hi) {
    v16h f;
    const float* p = row + kb + hi * 16;
#pragma unroll
    for (int j = 0; j < 16; ++j) f[j] = (_Float16)clamp5(p[j]);
    return f;
}

// B-matrix fragment (f16 32x16, K x N) with B = K_tile^T: lane owns column
// N = lane&15 (= K-row k0+N); lanes 0-15 hold K=kb+0..15 sequentially,
// lanes 16-31 hold K=kb+16..31 (dense analog of the SWMMAC B layout).
__device__ __forceinline__ v16h load_frag_b(const float* __restrict__ row, int kb, int hi) {
    v16h f;
    const float* p = row + kb + hi * 16;
#pragma unroll
    for (int j = 0; j < 16; ++j) f[j] = (_Float16)clamp5(p[j]);
    return f;
}

extern "C" __global__ __launch_bounds__(256)
void attn_map_kernel(const float* __restrict__ keys,
                     const float* __restrict__ queries,
                     const float* __restrict__ pos_key,
                     const float* __restrict__ pos_query,
                     const unsigned char* __restrict__ mask,
                     float* __restrict__ out)
{
    extern __shared__ float smem[];  // 16 x 2048 f32 scores = 128 KB

    const int tid  = threadIdx.x;
    const int wave = tid >> 5;
    const int lane = tid & 31;
    const int r16  = lane & 15;
    const int hi   = lane >> 4;

    const int bid = blockIdx.x;        // B*H*(S/16) blocks
    const int qt  = bid & 127;         // q tile within head
    const int bh  = bid >> 7;          // b*H + h
    const int b   = bh >> 4;
    const int q0  = qt * 16;

    const size_t baseQK  = (size_t)bh * S_LEN * DHEAD;
    const size_t basePos = (size_t)bh * S_LEN * PDHEAD;

    // -------- per-wave A fragments (held in VGPRs for all 16 k-tiles) -------
    const float* qrow  = queries   + baseQK  + (size_t)(q0 + r16) * DHEAD;
    const float* pqrow = pos_query + basePos + (size_t)(q0 + r16) * PDHEAD;
    const v16h aq0 = load_frag_a(qrow,  0, hi);
    const v16h aq1 = load_frag_a(qrow, 32, hi);
    const v16h apq = load_frag_a(pqrow, 0, hi);

    // -------- phase 1: scores -> LDS (wave-uniform loop, EXEC stays full) ---
    for (int kt = wave; kt < S_LEN / 16; kt += 8) {
        const int k0 = kt * 16;
        const float* krow  = keys    + baseQK  + (size_t)(k0 + r16) * DHEAD;
        const float* pkrow = pos_key + basePos + (size_t)(k0 + r16) * PDHEAD;
        const v16h bk0 = load_frag_b(krow,  0, hi);
        const v16h bk1 = load_frag_b(krow, 32, hi);
        const v16h bpk = load_frag_b(pkrow, 0, hi);

        v8f c = {};
        c = __builtin_amdgcn_wmma_f32_16x16x32_f16(false, aq0, false, bk0,
                                                   (short)0, c, false, false);
        c = __builtin_amdgcn_wmma_f32_16x16x32_f16(false, aq1, false, bk1,
                                                   (short)0, c, false, false);
        v8f cp = {};
        cp = __builtin_amdgcn_wmma_f32_16x16x32_f16(false, apq, false, bpk,
                                                    (short)0, cp, false, false);

        const int  kcol   = k0 + r16;  // column N this lane owns in C/D
        const bool masked = mask[(size_t)b * S_LEN + kcol] != 0;
        const float ninf  = -__builtin_inff();
#pragma unroll
        for (int r = 0; r < 8; ++r) {
            const float s = masked ? ninf : (c[r] * SCALE + cp[r] * PSCALE);
            smem[(r + hi * 8) * S_LEN + kcol] = s;  // row M = r + hi*8
        }
    }

    __syncthreads();

    // -------- phase 2: row softmax; b128 LDS reads + b128 coalesced stores --
    for (int m = wave; m < 16; m += 8) {
        const v4f* rowp4 = (const v4f*)(smem + m * S_LEN);

        float mx = -__builtin_inff();
#pragma unroll
        for (int j = 0; j < S_LEN / 128; ++j) {        // 16 x ds_load_b128
            const v4f v = rowp4[j * 32 + lane];
            mx = fmaxf(mx, fmaxf(fmaxf(v[0], v[1]), fmaxf(v[2], v[3])));
        }
#pragma unroll
        for (int off = 16; off > 0; off >>= 1) mx = fmaxf(mx, __shfl_xor(mx, off, 32));

        v4f ex[S_LEN / 128];
        float sum = 0.0f;
#pragma unroll
        for (int j = 0; j < S_LEN / 128; ++j) {
            const v4f v = rowp4[j * 32 + lane];
            v4f e;
            e[0] = __expf(v[0] - mx);
            e[1] = __expf(v[1] - mx);
            e[2] = __expf(v[2] - mx);
            e[3] = __expf(v[3] - mx);
            ex[j] = e;
            sum += (e[0] + e[1]) + (e[2] + e[3]);
        }
#pragma unroll
        for (int off = 16; off > 0; off >>= 1) sum += __shfl_xor(sum, off, 32);
        const float inv = 1.0f / sum;

        v4f* orow4 = (v4f*)(out + ((size_t)bh * S_LEN + (size_t)(q0 + m)) * S_LEN);
#pragma unroll
        for (int j = 0; j < S_LEN / 128; ++j) {        // 16 x global_store_b128
            v4f e = ex[j];
            e[0] *= inv; e[1] *= inv; e[2] *= inv; e[3] *= inv;
            orow4[j * 32 + lane] = e;
        }
    }
}

extern "C" void kernel_launch(void* const* d_in, const int* in_sizes, int n_in,
                              void* d_out, int out_size, void* d_ws, size_t ws_size,
                              hipStream_t stream) {
    const float* keys      = (const float*)d_in[0];
    const float* queries   = (const float*)d_in[1];
    const float* pos_key   = (const float*)d_in[2];
    const float* pos_query = (const float*)d_in[3];
    const unsigned char* mask = (const unsigned char*)d_in[4];
    float* out = (float*)d_out;

    const size_t lds_bytes = (size_t)16 * S_LEN * sizeof(float);  // 128 KB
    // gfx1250 WGP has 320 KB LDS; raise the per-kernel dynamic-LDS cap.
    hipFuncSetAttribute((const void*)attn_map_kernel,
                        hipFuncAttributeMaxDynamicSharedMemorySize,
                        (int)lds_bytes);

    dim3 grid(NBATCH * NHEAD * (S_LEN / 16));  // 4096 blocks
    dim3 block(256);                           // 8 wave32 waves
    attn_map_kernel<<<grid, block, lds_bytes, stream>>>(
        keys, queries, pos_key, pos_query, mask, out);
}
